// RNN_15144054686346
// MI455X (gfx1250) — matrix operand order repaired and checked by reading, not compiled, not run on hardware
//
#include <hip/hip_runtime.h>
#include <hip/hip_bf16.h>
#include <math.h>

// Problem dims (fixed by the reference)
#define BB 64
#define SS 512
#define EE 512
#define HH 1024
#define CC 5
#define MM (SS * BB)          // 32768 rows of the input-projection GEMM
#define NBLK 32               // persistent blocks (each owns 32 columns of U)
#define LSTRIDE (HH + 8)      // padded LDS row stride (bank-conflict-free frags)

typedef __attribute__((ext_vector_type(16))) __bf16 v16bf;
typedef __attribute__((ext_vector_type(8)))  float  v8f;
typedef int v4i_async __attribute__((vector_size(16)));   // builtin's pointee type

union Frag16 { v16bf v; uint4 q[2]; };

__device__ __forceinline__ unsigned short f2bf_bits(float f) {
    union { float f; unsigned u; } x; x.f = f;
    unsigned r = x.u + 0x7FFFu + ((x.u >> 16) & 1u);   // round-to-nearest-even
    return (unsigned short)(r >> 16);
}
__device__ __forceinline__ float bf2f(unsigned short b) {
    union { unsigned u; float f; } x; x.u = ((unsigned)b) << 16;
    return x.f;
}

// ---------------------------------------------------------------- utilities
__global__ __launch_bounds__(256) void cvt_bf16(const float* __restrict__ src,
                                                unsigned short* __restrict__ dst, int n) {
    int i = blockIdx.x * blockDim.x + threadIdx.x;
    if (i < n) dst[i] = f2bf_bits(src[i]);
}

__global__ __launch_bounds__(256) void zero_u16(unsigned short* __restrict__ p, int n) {
    int i = blockIdx.x * blockDim.x + threadIdx.x;
    if (i < n) p[i] = 0;
}

__global__ void zero_u32(unsigned int* __restrict__ p) { p[0] = 0u; }

// embA[m][e] = bf16(emb[x[b][t]][e]), m = t*B + b  (xs layout is [S,B,H])
__global__ __launch_bounds__(256) void gather_embed(const int* __restrict__ x,
                                                    const float* __restrict__ emb,
                                                    unsigned short* __restrict__ embA) {
    int gid = blockIdx.x * blockDim.x + threadIdx.x;   // MM*EE threads
    int m = gid >> 9;          // /512
    int e = gid & 511;
    int t = m >> 6;            // /64
    int b = m & 63;
    int tok = x[b * SS + t];
    embA[gid] = f2bf_bits(emb[(size_t)tok * EE + e]);
}

// ------------------------------------------- fragment loaders (bf16 16x16x32)
// A 16x32: lane<16 -> row m0+lane, K k0..k0+15 ; lane>=16 -> K k0+16..k0+31
__device__ __forceinline__ v16bf load_fragA(const unsigned short* __restrict__ base,
                                            int stride, int m0, int k0, int lane) {
    Frag16 f;
    const unsigned short* p = base + (size_t)(m0 + (lane & 15)) * stride
                                   + k0 + ((lane >> 4) << 4);
    f.q[0] = *(const uint4*)p;
    f.q[1] = *(const uint4*)(p + 8);
    return f.v;
}
// B 32x16 from a row-major [N,K] matrix (computes A @ Mat^T):
// lane<16 -> col n0+lane, K k0..k0+15 ; lane>=16 -> K k0+16..k0+31 (contig 32B)
__device__ __forceinline__ v16bf load_fragBT(const unsigned short* __restrict__ base,
                                             int stride, int n0, int k0, int lane) {
    Frag16 f;
    const unsigned short* p = base + (size_t)(n0 + (lane & 15)) * stride
                                   + k0 + ((lane >> 4) << 4);
    f.q[0] = *(const uint4*)p;
    f.q[1] = *(const uint4*)(p + 8);
    return f.v;
}
// Same pattern but from LDS (generic pointer into __shared__ -> ds_load_b128)
__device__ __forceinline__ v16bf load_fragBT_lds(const unsigned short* base,
                                                 int n0, int k0, int lane) {
    Frag16 f;
    const unsigned short* p = base + (n0 + (lane & 15)) * LSTRIDE
                                   + k0 + ((lane >> 4) << 4);
    f.q[0] = *(const uint4*)p;
    f.q[1] = *(const uint4*)(p + 8);
    return f.v;
}

// -------------------------------------------------- xs = embA @ W^T + bW
// M=32768, K=512, N=1024. One wave per 16x16 tile; 131072 tiles; 8 waves/block.
__global__ __launch_bounds__(256) void xs_gemm(const unsigned short* __restrict__ embA,
                                               const unsigned short* __restrict__ Wb,
                                               const float* __restrict__ bW,
                                               unsigned short* __restrict__ xs) {
    const int lane = threadIdx.x & 31;
    const int wave = threadIdx.x >> 5;
    const int tile = blockIdx.x * 8 + wave;
    const int tn = tile & 63;          // N/16 = 64
    const int tm = tile >> 6;          // 0..2047
    const int m0 = tm << 4, n0 = tn << 4;
    const int col = lane & 15, hi = lane >> 4;

    v8f c;
    float bias = bW[n0 + col];
#pragma unroll
    for (int r = 0; r < 8; ++r) c[r] = bias;   // preload bias into accumulator

    for (int k0 = 0; k0 < EE; k0 += 32) {
        v16bf a = load_fragA (embA, EE, m0, k0, lane);
        v16bf b = load_fragBT(Wb,   EE, n0, k0, lane);
        c = __builtin_amdgcn_wmma_f32_16x16x32_bf16(false, a, false, b,
                                                    (short)0, c, false, false);
    }
    unsigned short* out = xs + (size_t)m0 * HH + n0 + col;
#pragma unroll
    for (int r = 0; r < 8; ++r)
        out[(size_t)(r + (hi << 3)) * HH] = f2bf_bits(c[r]);
}

// ------------------------------------------------------------------------
// Persistent recurrence kernel: 32 blocks x 8 waves, block b owns U columns
// [32b, 32b+32). U slice staged into LDS once via async-to-LDS, then 512
// steps of  h = tanh(xs_t + h @ U^T)  with a device-scope grid barrier
// between steps (monotonic counter, no reset races).
// ------------------------------------------------------------------------
__global__ __launch_bounds__(256) void rnn_persistent(
        const unsigned short* __restrict__ xs,    // [S][64][1024] bf16
        const unsigned short* __restrict__ Ub,    // [1024][1024]  bf16
        unsigned short* __restrict__ hA,          // ping  (holds h0 = 0)
        unsigned short* __restrict__ hB,          // pong
        unsigned int*   __restrict__ barCnt) {
    __shared__ unsigned short Ulds[32 * LSTRIDE];

    const int lane = threadIdx.x & 31;
    const int wave = threadIdx.x >> 5;
    const int blk  = blockIdx.x;                  // 0..31
    const int col  = lane & 15, hi = lane >> 4;

    // ---- stage this block's 32 U rows (64 KB) into LDS, 16B granules ----
    for (int idx = threadIdx.x; idx < 32 * (HH / 8); idx += 256) {
        int row   = idx >> 7;                     // 128 16B-chunks per row
        int chunk = idx & 127;
        const uint4* src = (const uint4*)(Ub + (size_t)(32 * blk + row) * HH + chunk * 8);
        uint4*       dst = (uint4*)(&Ulds[row * LSTRIDE + chunk * 8]);
#if __has_builtin(__builtin_amdgcn_global_load_async_to_lds_b128)
        __builtin_amdgcn_global_load_async_to_lds_b128(
            (__attribute__((address_space(1))) v4i_async*)src,
            (__attribute__((address_space(3))) v4i_async*)dst, 0, 0);
#else
        *dst = *src;                              // global_load + ds_store fallback
#endif
    }
#if __has_builtin(__builtin_amdgcn_global_load_async_to_lds_b128)
#if __has_builtin(__builtin_amdgcn_s_wait_asynccnt)
    __builtin_amdgcn_s_wait_asynccnt(0);
#else
    asm volatile("s_wait_asynccnt 0" ::: "memory");
#endif
#endif
    __syncthreads();

    // wave -> 16x16 tile: m0 in {0,16,32,48}, n_local in {0,16}
    const int m0   = (wave & 3) << 4;
    const int nloc = (wave >> 2) << 4;
    const int n0   = (blk << 5) + nloc;           // global column base

    const unsigned short* hin  = hA;
    unsigned short*       hout = hB;

    for (int t = 0; t < SS; ++t) {
        const unsigned short* xsT = xs + (size_t)t * (BB * HH);

        v8f c;
#pragma unroll
        for (int r = 0; r < 8; ++r)               // preload xs_t tile into C
            c[r] = bf2f(xsT[(size_t)(m0 + r + (hi << 3)) * HH + n0 + col]);

        for (int k0 = 0; k0 < HH; k0 += 32) {
            v16bf a = load_fragA(hin, HH, m0, k0, lane);     // global (L2)
            v16bf b = load_fragBT_lds(Ulds, nloc, k0, lane); // LDS
            c = __builtin_amdgcn_wmma_f32_16x16x32_bf16(false, a, false, b,
                                                        (short)0, c, false, false);
        }
        unsigned short* out = hout + (size_t)m0 * HH + n0 + col;
#pragma unroll
        for (int r = 0; r < 8; ++r)
            out[(size_t)(r + (hi << 3)) * HH] = f2bf_bits(tanhf(c[r]));

        // ---- device-scope grid barrier (monotonic counter) ----
        __threadfence();                          // release h stores (agent scope)
        __syncthreads();
        if (threadIdx.x == 0) {
            __hip_atomic_fetch_add(barCnt, 1u, __ATOMIC_RELEASE,
                                   __HIP_MEMORY_SCOPE_AGENT);
            const unsigned int target = (unsigned int)(t + 1) * NBLK;
            while (__hip_atomic_load(barCnt, __ATOMIC_ACQUIRE,
                                     __HIP_MEMORY_SCOPE_AGENT) < target)
                __builtin_amdgcn_s_sleep(2);
        }
        __syncthreads();
        __threadfence();                          // acquire: see other blocks' h

        const unsigned short* tmp = hout; hout = (unsigned short*)hin; hin = tmp;
    }
    // after 512 (even) steps the final h sits in hA
}

// ------------------------------------------------------------ epilogues
__global__ __launch_bounds__(256) void final_h(const unsigned short* __restrict__ hA,
                                               float* __restrict__ hOut) {
    int i = blockIdx.x * blockDim.x + threadIdx.x;   // 65536 threads
    hOut[i] = bf2f(hA[i]);
}

__global__ __launch_bounds__(320) void out_proj(const unsigned short* __restrict__ hA,
                                                const float* __restrict__ V,
                                                const float* __restrict__ bV,
                                                float* __restrict__ out) {
    int i = threadIdx.x;                // 320 = B*C
    int b = i / CC, cIdx = i % CC;
    float s = bV[cIdx];
    for (int k = 0; k < HH; ++k)
        s += bf2f(hA[b * HH + k]) * V[cIdx * HH + k];
    out[i] = s;
}

// ------------------------------------------------------------ launcher
extern "C" void kernel_launch(void* const* d_in, const int* in_sizes, int n_in,
                              void* d_out, int out_size, void* d_ws, size_t ws_size,
                              hipStream_t stream) {
    const int*   x   = (const int*)  d_in[0];
    const float* emb = (const float*)d_in[1];
    const float* W   = (const float*)d_in[2];
    const float* bW  = (const float*)d_in[3];
    const float* U   = (const float*)d_in[4];
    const float* V   = (const float*)d_in[5];
    const float* bV  = (const float*)d_in[6];
    float* out = (float*)d_out;

    char* ws = (char*)d_ws;
    unsigned short* embA = (unsigned short*)(ws + 0);                     // 32 MB
    unsigned short* xs   = (unsigned short*)(ws + 33554432ull);           // 64 MB
    unsigned short* Ub   = (unsigned short*)(ws + 100663296ull);          // 2 MB
    unsigned short* Wb   = (unsigned short*)(ws + 102760448ull);          // 1 MB
    unsigned short* hA   = (unsigned short*)(ws + 103809024ull);          // 128 KB
    unsigned short* hB   = (unsigned short*)(ws + 103940096ull);          // 128 KB
    unsigned int*   cnt  = (unsigned int*)  (ws + 104071168ull);          // 4 B

    cvt_bf16    <<<(HH * EE) / 256, 256, 0, stream>>>(W, Wb, HH * EE);
    cvt_bf16    <<<(HH * HH) / 256, 256, 0, stream>>>(U, Ub, HH * HH);
    gather_embed<<<(MM * EE) / 256, 256, 0, stream>>>(x, emb, embA);
    zero_u16    <<<(BB * HH) / 256, 256, 0, stream>>>(hA, BB * HH);
    zero_u32    <<<1, 1, 0, stream>>>(cnt);

    // xs = embA @ W^T + bW   (2048 x 64 tiles, 8 waves/block)
    xs_gemm<<<(MM / 16) * (HH / 16) / 8, 256, 0, stream>>>(embA, Wb, bW, xs);

    // entire 512-step recurrence in one persistent kernel
    rnn_persistent<<<NBLK, 256, 0, stream>>>(xs, Ub, hA, hB, cnt);

    final_h <<<(BB * HH) / 256, 256, 0, stream>>>(hA, out + BB * CC);
    out_proj<<<1, BB * CC, 0, stream>>>(hA, V, bV, out);
}
